// XCY_44375602102939
// MI455X (gfx1250) — matrix operand order repaired and checked by reading, not compiled, not run on hardware
//
#include <hip/hip_runtime.h>
#include <hip/hip_bf16.h>
#include <math.h>

// ---------------------------------------------------------------------------
// MI455X (gfx1250, wave32) implementation.
//  K1: per-token inverse norms
//  K2: LDS-tiled transpose -> token-major fp32 Xt, bf16 Xa (raw a rows),
//      bf16 Xbn (b rows pre-scaled by 1/||b||)
//  K3: sim = Xa x Xbn^T via v_wmma_f32_16x16x32_bf16 + fused argmax
//  K4: fused-row build (closed-form spa partner) + 1x1 conv via
//      v_wmma_f32_16x16x4_f32 + BN + SiLU
// ---------------------------------------------------------------------------

typedef __attribute__((ext_vector_type(16))) __bf16 v16bf;
typedef __attribute__((ext_vector_type(8)))  __bf16 v8bf;
typedef __attribute__((ext_vector_type(8)))  float  v8f;
typedef __attribute__((ext_vector_type(2)))  float  v2f;

#define B_   16
#define C_   256
#define T_   4096
#define NA_  1024
#define NB_  3072
#define O_   512
#define HW_  1024   // 32*32 output spatial

__device__ __forceinline__ unsigned short f2bf_bits(float f) {
    unsigned int u = __float_as_uint(f);
    u += 0x7fffu + ((u >> 16) & 1u);      // round-to-nearest-even
    return (unsigned short)(u >> 16);
}

// ---------------- K1: inverse token norms (coalesced along tokens) ---------
__global__ void knorm(const float* __restrict__ x, float* __restrict__ invn) {
    int tid = blockIdx.x * blockDim.x + threadIdx.x;   // b*T_ + tok
    int b = tid >> 12, tok = tid & (T_ - 1);
    const float* p = x + (size_t)b * C_ * T_ + tok;
    float s = 0.f;
#pragma unroll 8
    for (int c = 0; c < C_; ++c) {
        float v = p[(size_t)c * T_];
        s += v * v;
    }
    invn[tid] = rsqrtf(s);
}

// ---------------- K2: transpose + pack (coalesced read AND write) ----------
__global__ void kpack(const float* __restrict__ x, const float* __restrict__ invn,
                      float* __restrict__ Xt, unsigned short* __restrict__ Xa,
                      unsigned short* __restrict__ Xb) {
    __shared__ float tile[64][65];                 // padded: conflict-free both ways
    int b = blockIdx.z, c0 = blockIdx.y * 64, t0 = blockIdx.x * 64;
    int tx = threadIdx.x, ty = threadIdx.y;        // block (64,4)
    for (int cc = ty; cc < 64; cc += 4)
        tile[cc][tx] = x[((size_t)b * C_ + c0 + cc) * T_ + t0 + tx];
    __syncthreads();
    for (int j = ty; j < 64; j += 4) {
        int flat = t0 + j;
        float v = tile[tx][j];                     // channel c0+tx of token flat
        Xt[((size_t)b * T_ + flat) * C_ + c0 + tx] = v;
        if ((flat & 3) == 0) {
            Xa[((size_t)b * NA_ + (flat >> 2)) * C_ + c0 + tx] = f2bf_bits(v);
        } else {
            int bpos = flat - (flat >> 2) - 1;     // position in b_idx ordering
            float vn = v * invn[b * T_ + flat];    // pre-scale by 1/||b||
            Xb[((size_t)b * NB_ + bpos) * C_ + c0 + tx] = f2bf_bits(vn);
        }
    }
}

// ---------------- K3: cosine-sim GEMM (bf16 WMMA) + argmax -----------------
__global__ void __launch_bounds__(256, 2)
ksim(const unsigned short* __restrict__ XaU, const unsigned short* __restrict__ XbU,
     int* __restrict__ simflat) {
    int atile = blockIdx.x, b = blockIdx.y;
    int lane = threadIdx.x & 31, wave = threadIdx.x >> 5;
    int half = lane >> 4, m = lane & 15;
    __shared__ float redv[8][16];
    __shared__ int   redi[8][16];

    // A fragments (16 a-rows x K=256), loaded once, kept in VGPRs.
    // A layout (matches B): lanes<16 hold K {0..15}, lanes>=16 {16..31}.
    const __bf16* aRow = (const __bf16*)XaU + ((size_t)b * NA_ + atile * 16 + m) * C_;
    v16bf afrag[8];
#pragma unroll
    for (int s = 0; s < 8; ++s) {
        v8bf p0 = *(const v8bf*)(aRow + 32 * s + 16 * half);
        v8bf p1 = *(const v8bf*)(aRow + 32 * s + 16 * half + 8);
        union { v16bf v; v8bf h[2]; } u;
        u.h[0] = p0; u.h[1] = p1;
        afrag[s] = u.v;
    }

    float bestv[8]; int besti[8];
#pragma unroll
    for (int r = 0; r < 8; ++r) { bestv[r] = -3.4e38f; besti[r] = 0; }

    const __bf16* bBase = (const __bf16*)XbU + (size_t)b * NB_ * C_;
    for (int nt = wave; nt < NB_ / 16; nt += 8) {
        // B layout (matches A): lane holds column n, 16 contiguous K values.
        const __bf16* bRow = bBase + ((size_t)nt * 16 + m) * C_;
        v8f acc = {};
#pragma unroll
        for (int s = 0; s < 8; ++s) {
            v16bf bfrag = *(const v16bf*)(bRow + 32 * s + 16 * half);
            acc = __builtin_amdgcn_wmma_f32_16x16x32_bf16(
                false, afrag[s], false, bfrag, (short)0, acc, false, false);
        }
        int nidx = nt * 16 + m;                    // b-position of this lane's column
#pragma unroll
        for (int r = 0; r < 8; ++r)                // strict > keeps lowest index on tie
            if (acc[r] > bestv[r]) { bestv[r] = acc[r]; besti[r] = nidx; }
    }

    // cross-lane argmax within each 16-lane half (rows 0-7 / 8-15)
#pragma unroll
    for (int mask = 8; mask >= 1; mask >>= 1) {
#pragma unroll
        for (int r = 0; r < 8; ++r) {
            float ov = __shfl_xor(bestv[r], mask, 32);
            int   oi = __shfl_xor(besti[r], mask, 32);
            if (ov > bestv[r] || (ov == bestv[r] && oi < besti[r])) { bestv[r] = ov; besti[r] = oi; }
        }
    }
    if ((lane & 15) == 0) {
#pragma unroll
        for (int r = 0; r < 8; ++r) { redv[wave][r + 8 * half] = bestv[r]; redi[wave][r + 8 * half] = besti[r]; }
    }
    __syncthreads();
    if (threadIdx.x < 16) {
        int row = threadIdx.x;
        float bv = redv[0][row]; int bi = redi[0][row];
        for (int w = 1; w < 8; ++w) {
            float ov = redv[w][row]; int oi = redi[w][row];
            if (ov > bv || (ov == bv && oi < bi)) { bv = ov; bi = oi; }
        }
        // map b-position -> flat token index: per row 48 b-cols, cols skip %4==0
        int rr = bi / 48, c48 = bi % 48;
        int col = c48 + c48 / 3 + 1;
        simflat[b * NA_ + atile * 16 + row] = rr * 64 + col;
    }
}

// ---------------- K4: fuse + 1x1 conv (fp32 WMMA) + BN + SiLU --------------
__global__ void __launch_bounds__(256, 2)
kfuseconv(const float* __restrict__ Xt, const int* __restrict__ simflat,
          const float* __restrict__ Wc, const float* __restrict__ gamma,
          const float* __restrict__ beta, const float* __restrict__ mean,
          const float* __restrict__ var, const float* __restrict__ fwts,
          float* __restrict__ out) {
    __shared__ float fused[16 * 264];              // 16 tokens x 256ch, padded stride
    int atile = blockIdx.x, b = blockIdx.y;

    float w0 = fminf(fmaxf(fwts[0], 0.f), 6.f);
    float w1 = fminf(fmaxf(fwts[1], 0.f), 6.f);
    float invw = 1.f / (w0 + w1 + 1e-8f);
    float fw0 = w0 * invw, fw1 = w1 * invw;

    int c = threadIdx.x;                           // 256 threads = 256 channels
    const float* base = Xt + (size_t)b * T_ * C_;
    for (int i = 0; i < 16; ++i) {
        int alin = atile * 16 + i;
        int aflat = alin * 4;
        int col = aflat & 63;
        int spa = (col == 0) ? aflat + 1 : aflat - 1;   // closed-form spatial partner
        int sel = simflat[b * NA_ + alin];              // broadcast load
        float xa = base[(size_t)aflat * C_ + c];
        float xs = base[(size_t)sel  * C_ + c];
        float xp = base[(size_t)spa  * C_ + c];
        fused[i * 264 + c] = 0.5f * ((fw0 + fw1) * xa + fw0 * xs + fw1 * xp);
    }
    __syncthreads();

    int lane = threadIdx.x & 31, wave = threadIdx.x >> 5;
    int half = lane >> 4, n = lane & 15;
    for (int q = 0; q < 4; ++q) {                  // 8 waves x 4 tiles = 512 out-ch
        int o = (wave * 4 + q) * 16 + n;
        const float* wrow = Wc + (size_t)o * C_;
        v8f acc = {};
        // f32 A 16x4: lane m=n, VGPR0/1 = K {2*half, 2*half+1};  B 4x16 symmetric.
#pragma unroll 4
        for (int k0 = 0; k0 < C_; k0 += 4) {
            v2f a  = *(const v2f*)&fused[n * 264 + k0 + 2 * half];
            v2f bb = *(const v2f*)(wrow + k0 + 2 * half);
            acc = __builtin_amdgcn_wmma_f32_16x16x4_f32(
                false, a, false, bb, (short)0, acc, false, false);
        }
        float sc = gamma[o] * rsqrtf(var[o] + 1e-5f);
        float sh = beta[o] - mean[o] * sc;
        float res[8];
#pragma unroll
        for (int r = 0; r < 8; ++r) {              // D: VGPR r -> token r+8*half
            float y = acc[r] * sc + sh;
            res[r] = y / (1.f + __expf(-y));       // SiLU
        }
        float* op = out + ((size_t)b * O_ + o) * HW_ + atile * 16 + 8 * half;
        *(float4*)(op)     = make_float4(res[0], res[1], res[2], res[3]);
        *(float4*)(op + 4) = make_float4(res[4], res[5], res[6], res[7]);
    }
}

// ---------------------------------------------------------------------------
extern "C" void kernel_launch(void* const* d_in, const int* in_sizes, int n_in,
                              void* d_out, int out_size, void* d_ws, size_t ws_size,
                              hipStream_t stream) {
    const float* x    = (const float*)d_in[0];
    const float* Wc   = (const float*)d_in[1];
    const float* gmm  = (const float*)d_in[2];
    const float* bta  = (const float*)d_in[3];
    const float* mn   = (const float*)d_in[4];
    const float* vr   = (const float*)d_in[5];
    const float* fwts = (const float*)d_in[6];
    float* out = (float*)d_out;

    // workspace layout (all 256B-aligned):
    char* w = (char*)d_ws;
    float*          invn    = (float*)(w + 0);          //  256 KB
    int*            simflat = (int*)  (w + 262144);     //   64 KB
    unsigned short* Xa      = (unsigned short*)(w + 327680);    //  8 MB bf16
    unsigned short* Xb      = (unsigned short*)(w + 8716288);   // 24 MB bf16
    float*          Xt      = (float*)(w + 33882112);           // 64 MB fp32 token-major

    knorm    <<<B_ * T_ / 256, 256, 0, stream>>>(x, invn);
    kpack    <<<dim3(T_ / 64, C_ / 64, B_), dim3(64, 4), 0, stream>>>(x, invn, Xt, Xa, Xb);
    ksim     <<<dim3(NA_ / 16, B_), 256, 0, stream>>>(Xa, Xb, simflat);
    kfuseconv<<<dim3(NA_ / 16, B_), 256, 0, stream>>>(Xt, simflat, Wc, gmm, bta, mn, vr, fwts, out);
}
